// CrossAttentionModule_33621003993943
// MI455X (gfx1250) — matrix operand order repaired and checked by reading, not compiled, not run on hardware
//
#include <hip/hip_runtime.h>
#include <hip/hip_bf16.h>

typedef __attribute__((ext_vector_type(16))) _Float16 v16h;
typedef __attribute__((ext_vector_type(8)))  _Float16 v8h;
typedef __attribute__((ext_vector_type(8)))  float    v8f;

#define HID 128
#define NHEADS 4
#define INV_SQRT_DH 0.17677669529663687f  // 1/sqrt(32)

// ---------------------------------------------------------------------------
// helpers
// ---------------------------------------------------------------------------

// B fragment (32x16, f16) for v_wmma_f32_16x16x32_f16.
// Layout (this code's order, contiguous K per half-wave, same as lds_a_frag):
//   lane L: col n (fixed), halves j=0..15 hold K = kb + j,
//   kb = kstep*32 + (lane>=16 ? 16 : 0).  W row-major [K][N], f32->f16.
__device__ inline v16h load_b_frag(const float* __restrict__ W, int n, int kb) {
  v16h f;
#pragma unroll
  for (int j = 0; j < 16; ++j) f[j] = (_Float16)W[(size_t)(kb + j) * HID + n];
  return f;
}

// A fragment (16x32, f16) from an LDS tile stored row-major [16][HID] halves.
// lane L: row = L&15; koff = (L>=16)?16:0;
//   halves 0..7  = K = kstep*32 + koff + 0..7
//   halves 8..15 = K = kstep*32 + koff + 8 + 0..7
__device__ inline v16h lds_a_frag(const _Float16* lds, int lane, int kstep) {
  int r = lane & 15;
  int koff = (lane & 16) ? 16 : 0;
  const _Float16* p = lds + r * HID + kstep * 32 + koff;
  v8h g0 = *(const v8h*)p;
  v8h g1 = *(const v8h*)(p + 8);
  v16h a;
#pragma unroll
  for (int j = 0; j < 8; ++j) { a[j] = g0[j]; a[j + 8] = g1[j]; }
  return a;
}

// 256 threads cooperatively load a 16x128 f32 tile, convert to f16 in LDS.
__device__ inline void load_tile_f16(const float* __restrict__ src,
                                     _Float16* dst, int row0) {
  int t = threadIdx.x;
  int r = t >> 4;            // 16 rows
  int c = (t & 15) * 8;      // 8 elems per thread
  const float4* s = (const float4*)(src + (size_t)(row0 + r) * HID + c);
  float4 x = s[0], y = s[1];
  _Float16* d = dst + r * HID + c;
  d[0] = (_Float16)x.x; d[1] = (_Float16)x.y; d[2] = (_Float16)x.z; d[3] = (_Float16)x.w;
  d[4] = (_Float16)y.x; d[5] = (_Float16)y.y; d[6] = (_Float16)y.z; d[7] = (_Float16)y.w;
}

__device__ inline v8f wmma_f16(v16h a, v16h b, v8f c) {
  return __builtin_amdgcn_wmma_f32_16x16x32_f16(false, a, false, b, (short)0, c,
                                                false, false);
}

// ---------------------------------------------------------------------------
// kernel 0: zero accumulators
// ---------------------------------------------------------------------------
__global__ void k_zero(float* p, int n) {
  int i = blockIdx.x * blockDim.x + threadIdx.x;
  if (i < n) p[i] = 0.0f;
}

// ---------------------------------------------------------------------------
// kernel 1: task_head[B][128] = task_embed @ W_task + b_task   (WMMA)
// ---------------------------------------------------------------------------
__global__ __launch_bounds__(256) void k_task_proj(
    const float* __restrict__ te, const float* __restrict__ W,
    const float* __restrict__ bias, float* __restrict__ th, int Brows) {
  __shared__ _Float16 Atile[16 * HID];
  int lane = threadIdx.x & 31;
  int wv = threadIdx.x >> 5;             // 8 waves -> 8 column tiles of 16
  int n = wv * 16 + (lane & 15);
  int kbh = (lane & 16) ? 16 : 0;
  v16h Bf[4];
#pragma unroll
  for (int kk = 0; kk < 4; ++kk) Bf[kk] = load_b_frag(W, n, kk * 32 + kbh);
  float bn = bias[n];
  int mbase = (lane & 16) ? 8 : 0;
  float* const outp = th + mbase * HID + n;   // loop-invariant per-lane base
  int ntiles = Brows >> 4;
  for (int tile = blockIdx.x; tile < ntiles; tile += gridDim.x) {
    int row0 = tile << 4;
    __syncthreads();
    load_tile_f16(te, Atile, row0);
    __syncthreads();
    v8f c = {};
#pragma unroll
    for (int kk = 0; kk < 4; ++kk) c = wmma_f16(lds_a_frag(Atile, lane, kk), Bf[kk], c);
    float* q = outp + (size_t)row0 * HID;     // one 64-bit add per tile
#pragma unroll
    for (int j = 0; j < 8; ++j) q[j * HID] = c[j] + bn;  // imm-offset stores
  }
}

// ---------------------------------------------------------------------------
// kernel 2: fg_head GEMM (WMMA) fused with score dot-products + segment sums
// ---------------------------------------------------------------------------
__global__ __launch_bounds__(256) void k_scores(
    const float* __restrict__ fg, const int* __restrict__ bidx,
    const float* __restrict__ W, const float* __restrict__ bias,
    const float* __restrict__ th, float* __restrict__ scores,
    float* __restrict__ sum_scores, int N) {
  __shared__ _Float16 Atile[16 * HID];
  __shared__ int bT[16];
  __shared__ float P[16][8];            // per-row partial sums, per wave
  int lane = threadIdx.x & 31;
  int wv = threadIdx.x >> 5;
  int n = wv * 16 + (lane & 15);
  int kbh = (lane & 16) ? 16 : 0;
  v16h Bf[4];
#pragma unroll
  for (int kk = 0; kk < 4; ++kk) Bf[kk] = load_b_frag(W, n, kk * 32 + kbh);
  float bn = bias[n];
  int mbase = (lane & 16) ? 8 : 0;
  const float* const thp = th + n;      // gather base
  int ntiles = N >> 4;
  for (int tile = blockIdx.x; tile < ntiles; tile += gridDim.x) {
    int row0 = tile << 4;
    __syncthreads();
    load_tile_f16(fg, Atile, row0);
    if (threadIdx.x < 16) bT[threadIdx.x] = bidx[row0 + threadIdx.x];
    __syncthreads();
    v8f c = {};
#pragma unroll
    for (int kk = 0; kk < 4; ++kk) c = wmma_f16(lds_a_frag(Atile, lane, kk), Bf[kk], c);
    float acc[8];
#pragma unroll
    for (int j = 0; j < 8; ++j) {
      int row = mbase + j;
      float t = thp[(size_t)bT[row] * HID];      // gather task_head (sorted idx)
      float v = (c[j] + bn) * t;
      v += __shfl_xor(v, 8);                     // reduce the 16 cols of this wave
      v += __shfl_xor(v, 4);
      v += __shfl_xor(v, 2);
      v += __shfl_xor(v, 1);
      acc[j] = v;
    }
    if ((lane & 15) == 0) {
#pragma unroll
      for (int j = 0; j < 8; ++j) P[mbase + j][wv] = acc[j];
    }
    __syncthreads();
    if (threadIdx.x < 64) {
      int row = threadIdx.x >> 2, h = threadIdx.x & 3;
      float s = (P[row][2 * h] + P[row][2 * h + 1]) * INV_SQRT_DH;
      scores[(size_t)(row0 + row) * NHEADS + h] = s;
      atomicAdd(&sum_scores[(size_t)bT[row] * NHEADS + h], s);
    }
  }
}

// ---------------------------------------------------------------------------
// kernel 3: segment boundaries of sorted batch_idx -> counts (atomic-free)
// ---------------------------------------------------------------------------
__global__ void k_bounds(const int* __restrict__ bidx, int* firstA, int* lastA, int N) {
  int i = blockIdx.x * blockDim.x + threadIdx.x;
  if (i >= N) return;
  int b = bidx[i];
  if (i == 0 || bidx[i - 1] != b) firstA[b] = i;
  if (i == N - 1 || bidx[i + 1] != b) lastA[b] = i + 1;
}

// kernel 4: mean[b,h] = sum/max(count,1e-8)  (the reference's scatter_mean)
__global__ void k_mean(const float* __restrict__ sum_scores,
                       const int* __restrict__ firstA, const int* __restrict__ lastA,
                       float* __restrict__ meanv, int B) {
  int t = blockIdx.x * blockDim.x + threadIdx.x;
  if (t >= B * NHEADS) return;
  int b = t >> 2;
  float cnt = (float)(lastA[b] - firstA[b]);
  meanv[t] = sum_scores[t] / fmaxf(cnt, 1e-8f);
}

// ---------------------------------------------------------------------------
// kernel 5: e = exp(s - mean[b]); segment-sum e with wave-aggregated atomics
// ---------------------------------------------------------------------------
__global__ void k_exp(const int* __restrict__ bidx, const float* __restrict__ meanv,
                      float* __restrict__ sc, float* __restrict__ sum_exp, int N) {
  int i = blockIdx.x * blockDim.x + threadIdx.x;
  int lane = threadIdx.x & 31;
  bool valid = (i < N);
  int b = valid ? bidx[i] : -1;
  int pb = __shfl_up(b, 1);
  bool leader = (lane == 0) || (pb != b);
#pragma unroll
  for (int h = 0; h < NHEADS; ++h) {
    float e = 0.0f;
    if (valid) {
      e = __expf(sc[(size_t)i * NHEADS + h] - meanv[(size_t)b * NHEADS + h]);
      sc[(size_t)i * NHEADS + h] = e;   // overwrite scores with exp (rewritten by k2 each launch)
    }
    float r = e;                        // segmented suffix-sum over equal-b runs (sorted)
#pragma unroll
    for (int off = 1; off < 32; off <<= 1) {
      float orv = __shfl_down(r, off);
      int ob = __shfl_down(b, off);
      if (lane + off < 32 && ob == b) r += orv;
    }
    if (leader && valid) atomicAdd(&sum_exp[(size_t)b * NHEADS + h], r);
  }
}

// ---------------------------------------------------------------------------
// kernel 6: value GEMM (WMMA) -> attn scaling -> W_out GEMM (WMMA) -> out
// ---------------------------------------------------------------------------
__global__ __launch_bounds__(256) void k_out(
    const float* __restrict__ fg, const int* __restrict__ bidx,
    const float* __restrict__ Wv, const float* __restrict__ bv,
    const float* __restrict__ Wo, const float* __restrict__ bo,
    const float* __restrict__ esc, const float* __restrict__ sum_exp,
    float* __restrict__ out, int N) {
  __shared__ _Float16 Atile[16 * HID];
  __shared__ _Float16 Vtile[16 * HID];
  __shared__ int bT[16];
  __shared__ float attnW[16][NHEADS];
  int lane = threadIdx.x & 31;
  int wv = threadIdx.x >> 5;
  int n = wv * 16 + (lane & 15);
  int h = wv >> 1;                      // 32-col head owning this wave's 16 cols
  int kbh = (lane & 16) ? 16 : 0;
  v16h Bv[4], Bo[4];
#pragma unroll
  for (int kk = 0; kk < 4; ++kk) {
    Bv[kk] = load_b_frag(Wv, n, kk * 32 + kbh);
    Bo[kk] = load_b_frag(Wo, n, kk * 32 + kbh);
  }
  float bvn = bv[n], bon = bo[n];
  int mbase = (lane & 16) ? 8 : 0;
  float* const outp = out + mbase * HID + n;        // loop-invariant bases
  _Float16* const vtp = Vtile + mbase * HID + n;
  const float* const awp = &attnW[mbase][h];
  int ntiles = N >> 4;
  for (int tile = blockIdx.x; tile < ntiles; tile += gridDim.x) {
    int row0 = tile << 4;
    __syncthreads();
    load_tile_f16(fg, Atile, row0);
    if (threadIdx.x < 16) bT[threadIdx.x] = bidx[row0 + threadIdx.x];
    __syncthreads();
    if (threadIdx.x < 64) {
      int row = threadIdx.x >> 2, hh = threadIdx.x & 3;
      int b = bT[row];
      attnW[row][hh] = esc[(size_t)(row0 + row) * NHEADS + hh] /
                       (sum_exp[(size_t)b * NHEADS + hh] + 1e-8f);
    }
    __syncthreads();
    v8f c = {};
#pragma unroll
    for (int kk = 0; kk < 4; ++kk) c = wmma_f16(lds_a_frag(Atile, lane, kk), Bv[kk], c);
#pragma unroll
    for (int j = 0; j < 8; ++j) {
      float v = (c[j] + bvn) * awp[j * NHEADS];
      vtp[j * HID] = (_Float16)v;       // restage scaled value_head as GEMM2's A
    }
    __syncthreads();
    v8f c2 = {};
#pragma unroll
    for (int kk = 0; kk < 4; ++kk) c2 = wmma_f16(lds_a_frag(Vtile, lane, kk), Bo[kk], c2);
    float* q = outp + (size_t)row0 * HID;           // one 64-bit add per tile
#pragma unroll
    for (int j = 0; j < 8; ++j) q[j * HID] = c2[j] + bon;  // imm-offset stores
  }
}

// ---------------------------------------------------------------------------
// launch
// ---------------------------------------------------------------------------
extern "C" void kernel_launch(void* const* d_in, const int* in_sizes, int n_in,
                              void* d_out, int out_size, void* d_ws, size_t ws_size,
                              hipStream_t stream) {
  const float* fg     = (const float*)d_in[0];
  const float* te     = (const float*)d_in[1];
  const int*   bidx   = (const int*)d_in[2];
  const float* W_fg   = (const float*)d_in[3];
  const float* b_fg   = (const float*)d_in[4];
  const float* W_task = (const float*)d_in[5];
  const float* b_task = (const float*)d_in[6];
  const float* W_val  = (const float*)d_in[7];
  const float* b_val  = (const float*)d_in[8];
  const float* W_out  = (const float*)d_in[9];
  const float* b_out  = (const float*)d_in[10];
  float* out = (float*)d_out;

  int N = in_sizes[0] / HID;   // 500000 (multiple of 16)
  int B = in_sizes[1] / HID;   // 16384  (multiple of 16)

  float* ws = (float*)d_ws;
  size_t o = 0;
  float* task_head  = ws + o; o += (size_t)B * HID;      // 8 MB
  float* scores     = ws + o; o += (size_t)N * NHEADS;   // 8 MB (scores -> exp)
  float* sum_scores = ws + o; o += (size_t)B * NHEADS;   // zeroed region start
  float* sum_exp    = ws + o; o += (size_t)B * NHEADS;
  int*   firstA     = (int*)(ws + o); o += B;
  int*   lastA      = (int*)(ws + o); o += B;
  float* meanv      = ws + o; o += (size_t)B * NHEADS;

  int zeroN = B * NHEADS * 2 + B * 2;  // sum_scores | sum_exp | firstA | lastA
  k_zero<<<(zeroN + 255) / 256, 256, 0, stream>>>(sum_scores, zeroN);

  k_task_proj<<<B / 16, 256, 0, stream>>>(te, W_task, b_task, task_head, B);
  k_scores<<<2048, 256, 0, stream>>>(fg, bidx, W_fg, b_fg, task_head, scores,
                                     sum_scores, N);
  k_bounds<<<(N + 255) / 256, 256, 0, stream>>>(bidx, firstA, lastA, N);
  k_mean<<<(B * NHEADS + 255) / 256, 256, 0, stream>>>(sum_scores, firstA, lastA,
                                                       meanv, B);
  k_exp<<<(N + 255) / 256, 256, 0, stream>>>(bidx, meanv, scores, sum_exp, N);
  k_out<<<2048, 256, 0, stream>>>(fg, bidx, W_val, b_val, W_out, b_out, scores,
                                  sum_exp, out, N);
}
